// TokenNMSW_17480516895338
// MI455X (gfx1250) — matrix operand order repaired and checked by reading, not compiled, not run on hardware
//
#include <hip/hip_runtime.h>
#include <hip/hip_bf16.h>

#define BATCH 4
#define NTOK  4096
#define DDIM  256
#define THRS  0.7f

typedef _Float16 v8h  __attribute__((ext_vector_type(8)));
typedef _Float16 v16h __attribute__((ext_vector_type(16)));
typedef float    v8f  __attribute__((ext_vector_type(8)));

static __device__ __forceinline__ unsigned wave_ballot32(bool p) {
#if __has_builtin(__builtin_amdgcn_ballot_w32)
    return __builtin_amdgcn_ballot_w32(p);
#else
    return (unsigned)__ballot(p);
#endif
}

// ---------------------------------------------------------------------------
// 1) Bitonic argsort (descending by score), one block of 1024 threads / batch.
//    LDS: 4096 keys + 4096 indices = 32 KB (well under 320 KB/WGP).
// ---------------------------------------------------------------------------
__global__ __launch_bounds__(1024) void nms_argsort(const float* __restrict__ scores,
                                                    int* __restrict__ order) {
    __shared__ float sk[NTOK];
    __shared__ int   sv[NTOK];
    const int b = blockIdx.x;
    const int tid = threadIdx.x;
    for (int t = tid; t < NTOK; t += 1024) { sk[t] = scores[b * NTOK + t]; sv[t] = t; }
    __syncthreads();
    for (int size = 2; size <= NTOK; size <<= 1) {
        for (int stride = size >> 1; stride > 0; stride >>= 1) {
            for (int t = tid; t < NTOK; t += 1024) {
                int j = t ^ stride;
                if (j > t) {
                    bool descBlock = ((t & size) == 0);
                    float a = sk[t], c = sk[j];
                    bool doSwap = descBlock ? (a < c) : (a > c);
                    if (doSwap) {
                        sk[t] = c; sk[j] = a;
                        int va = sv[t]; sv[t] = sv[j]; sv[j] = va;
                    }
                }
            }
            __syncthreads();
        }
    }
    for (int t = tid; t < NTOK; t += 1024) order[b * NTOK + t] = sv[t];
}

// ---------------------------------------------------------------------------
// 2) Normalize + gather into sorted order, fp32 -> f16. One wave per token.
// ---------------------------------------------------------------------------
__global__ __launch_bounds__(256) void nms_gather_norm(const float* __restrict__ tokens,
                                                       const int* __restrict__ order,
                                                       _Float16* __restrict__ t16) {
    const int wid  = (blockIdx.x * 256 + threadIdx.x) >> 5;  // global sorted slot
    const int lane = threadIdx.x & 31;
    const int b = wid >> 12;          // / 4096
    const int p = wid & (NTOK - 1);
    const int orig = order[b * NTOK + p];
    const float* src = tokens + ((size_t)b * NTOK + orig) * DDIM;
    float vals[8];
    float ss = 0.f;
#pragma unroll
    for (int q = 0; q < 8; ++q) { float x = src[lane + q * 32]; vals[q] = x; ss += x * x; }
#pragma unroll
    for (int off = 16; off > 0; off >>= 1) ss += __shfl_xor(ss, off, 32);
    const float inv = 1.0f / (sqrtf(ss) + 1e-6f);
    _Float16* dst = t16 + ((size_t)b * NTOK + p) * DDIM;
#pragma unroll
    for (int q = 0; q < 8; ++q) dst[lane + q * 32] = (_Float16)(vals[q] * inv);
}

// ---------------------------------------------------------------------------
// 3) WMMA similarity GEMM with ballot bit-packing of (sim > 0.7).
//    Each wave: 16 rows (tile-row it) x 128 cols (8 consecutive 16x16 tiles).
//    A fragment reused across the 8 column tiles. Strips strictly below the
//    diagonal are skipped (NMS only needs j > i).
//    adj layout: uint16 adj[b][row][NTOK/16]  (16 columns per word).
// ---------------------------------------------------------------------------
__global__ __launch_bounds__(256) void nms_simgemm(const _Float16* __restrict__ t16,
                                                   unsigned short* __restrict__ adj) {
    const int lane = threadIdx.x & 31;
    const int wave = threadIdx.x >> 5;
    const int b  = blockIdx.z;
    const int it = blockIdx.y;                  // row tile 0..255
    const int jg = blockIdx.x * 8 + wave;       // column strip 0..31 (8 tiles each)
    if (jg * 8 + 7 < it) return;                // entire strip below diagonal: skip

    const _Float16* tb = t16 + (size_t)b * NTOK * DDIM;
    const int half = lane >> 4;                 // 0 or 1
    const int lrow = lane & 15;

    const _Float16* arow = tb + (size_t)(it * 16 + lrow) * DDIM;
    v8f c[8] = {};

    for (int k0 = 0; k0 < DDIM; k0 += 32) {
        // A (16x32 f16): lanes 0-15 hold K[0..15], lanes 16-31 K[16..31]
        const _Float16* ap = arow + k0 + half * 16;
        v8h alo = *(const v8h*)(ap);
        v8h ahi = *(const v8h*)(ap + 8);
        v16h a = __builtin_shufflevector(alo, ahi, 0, 1, 2, 3, 4, 5, 6, 7,
                                                   8, 9, 10, 11, 12, 13, 14, 15);
#pragma unroll
        for (int q = 0; q < 8; ++q) {
            // B (32x16 f16): lane n = column n; lanes 0-15 K[0..15], 16-31 K[16..31]
            const _Float16* bp =
                tb + (size_t)((jg * 8 + q) * 16 + lrow) * DDIM + k0 + half * 16;
            v16h bf = *(const v16h*)bp;
            c[q] = __builtin_amdgcn_wmma_f32_16x16x32_f16(
                false, a, false, bf, (short)0, c[q], false, false);
        }
    }

    // Bit-pack: C VGPR r holds rows {r, r+8} x cols 0..15 across the 32 lanes,
    // so one ballot gives two 16-bit adjacency words.
#pragma unroll
    for (int q = 0; q < 8; ++q) {
        const int w = jg * 8 + q;  // column word index
#pragma unroll
        for (int r = 0; r < 8; ++r) {
            unsigned m = wave_ballot32(c[q][r] > THRS);
            if (lane == 0) {
                adj[((size_t)b * NTOK + it * 16 + r) * (NTOK / 16) + w] =
                    (unsigned short)(m & 0xFFFFu);
                adj[((size_t)b * NTOK + it * 16 + r + 8) * (NTOK / 16) + w] =
                    (unsigned short)(m >> 16);
            }
        }
    }
}

// ---------------------------------------------------------------------------
// 4) Greedy NMS scan. One wave per batch; the 4096-bit suppression set lives
//    in 4 VGPRs per lane (lane l owns bit columns [128l, 128l+128)).
// ---------------------------------------------------------------------------
static __device__ __forceinline__ unsigned maskgt(int i, int s) {
    // bits of a word covering global columns [s, s+31]; keep only columns > i
    if (i >= s + 31) return 0u;
    if (i < s) return ~0u;
    return (~0u) << (i - s + 1);
}

__global__ __launch_bounds__(32) void nms_scan(const unsigned short* __restrict__ adj,
                                               const int* __restrict__ order,
                                               float* __restrict__ keep_out) {
    const int b = blockIdx.x;
    const int lane = threadIdx.x;
    const unsigned short* ab = adj + (size_t)b * NTOK * (NTOK / 16);
    const int* ob = order + b * NTOK;
    float* kb = keep_out + b * NTOK;

    uint4 sup = {0u, 0u, 0u, 0u};
    for (int i = 0; i < NTOK; ++i) {
        const int owner = i >> 7;          // lane owning bit i
        const int sub = (i >> 5) & 3;      // word within that lane's uint4
        unsigned w = (sub == 0) ? sup.x : (sub == 1) ? sup.y : (sub == 2) ? sup.z : sup.w;
        unsigned wsel = __shfl(w, owner, 32);
        const bool kept = ((wsel >> (i & 31)) & 1u) == 0u;
        if (lane == 0) kb[ob[i]] = kept ? 1.0f : 0.0f;
        if (kept) {  // uniform branch: whole wave together
            const uint4 r = *(const uint4*)(ab + (size_t)i * (NTOK / 16) + lane * 8);
            const int s0 = lane * 128;
            sup.x |= r.x & maskgt(i, s0);
            sup.y |= r.y & maskgt(i, s0 + 32);
            sup.z |= r.z & maskgt(i, s0 + 64);
            sup.w |= r.w & maskgt(i, s0 + 96);
        }
    }
}

// ---------------------------------------------------------------------------
// 5) Apply keep mask to tokens (float4 streams).
// ---------------------------------------------------------------------------
__global__ __launch_bounds__(256) void nms_apply_mask(const float* __restrict__ tokens,
                                                      const float* __restrict__ keep,
                                                      float* __restrict__ out) {
    const size_t idx = (size_t)blockIdx.x * 256 + threadIdx.x;  // float4 index
    const float4* in4 = (const float4*)tokens;
    float4* out4 = (float4*)out;
    const float k = keep[idx >> 6];  // 64 float4 per token row (D=256)
    float4 v = in4[idx];
    out4[idx] = make_float4(v.x * k, v.y * k, v.z * k, v.w * k);
}

// ---------------------------------------------------------------------------
// Host launcher
// ---------------------------------------------------------------------------
extern "C" void kernel_launch(void* const* d_in, const int* in_sizes, int n_in,
                              void* d_out, int out_size, void* d_ws, size_t ws_size,
                              hipStream_t stream) {
    const float* tokens = (const float*)d_in[0];   // [4, 4096, 256] fp32
    const float* scores = (const float*)d_in[1];   // [4, 4096] fp32

    float* out_tokens = (float*)d_out;                                  // [4,4096,256]
    float* out_keep   = out_tokens + (size_t)BATCH * NTOK * DDIM;       // [4,4096]

    char* ws = (char*)d_ws;
    int* order           = (int*)ws;                                    // 64 KB
    _Float16* t16        = (_Float16*)(ws + 65536);                     // 8 MB
    unsigned short* adj  = (unsigned short*)(ws + 65536 + 8388608);     // 8 MB
    (void)in_sizes; (void)n_in; (void)out_size; (void)ws_size;

    nms_argsort<<<BATCH, 1024, 0, stream>>>(scores, order);
    nms_gather_norm<<<(BATCH * NTOK) / 8, 256, 0, stream>>>(tokens, order, t16);
    nms_simgemm<<<dim3(4, NTOK / 16, BATCH), 256, 0, stream>>>(t16, adj);
    nms_scan<<<BATCH, 32, 0, stream>>>(adj, order, out_keep);
    nms_apply_mask<<<(BATCH * NTOK * DDIM / 4) / 256, 256, 0, stream>>>(tokens, out_keep,
                                                                        out_tokens);
}
